// SE3FoldingTransformer_32220844654815
// MI455X (gfx1250) — matrix operand order repaired and checked by reading, not compiled, not run on hardware
//
#include <hip/hip_runtime.h>
#include <hip/hip_bf16.h>
#include <math.h>

typedef _Float16 half_t;
typedef __attribute__((ext_vector_type(16))) _Float16 v16h;
typedef __attribute__((ext_vector_type(8)))  _Float16 v8h;
typedef __attribute__((ext_vector_type(8)))  float    v8f;

#define BATCH 2
#define SEQ   1024
#define DIM   128
#define NH    4
#define DH    32
#define FFD   512
#define ROWS  (BATCH * SEQ)   // 2048
#define COORD_STEP 0.25f
#define LN_EPS 1e-5f

// ---------------------------------------------------------------------------
// WMMA GEMM: C = act(alpha * A(MxK,f16) * Bt + bias [+resid]) [*rowmask]
// B ALWAYS stored transposed: Bt[n*ldb + k]  -> contiguous 16B fragment loads.
// One wave computes a 16 x (16*NT) strip: A fragment loaded once per K-step,
// reused across NT WMMAs with independent accumulators.
// Batched via blockIdx.z: z -> (z/zdiv, z%zdiv) split strides.
// ---------------------------------------------------------------------------
template<int NT, int ACT>
__global__ __launch_bounds__(32)
void gemm_wmma_kernel(const half_t* A, int lda, long long sAo, long long sAi,
                      const half_t* B, int ldb, long long sBo, long long sBi,
                      const float* bias, float alpha,
                      const float* resid,
                      float* Cf, half_t* Ch,
                      int ldc, long long sCo, long long sCi,
                      const unsigned char* rowmask,
                      int zdiv, int K)
{
    const int tM = blockIdx.x, tN = blockIdx.y, z = blockIdx.z;
    const long long zo = z / zdiv, zi = z % zdiv;
    A += zo * sAo + zi * sAi;
    B += zo * sBo + zi * sBi;
    const long long coff = zo * sCo + zi * sCi;

    const int l  = threadIdx.x;
    const int lh = l & 15;
    const int hi = l >> 4;          // lane half: 0 or 1
    const int rowA = tM * 16 + lh;
    const int kbA = hi * 16;        // A: lanes 16-31 carry K offset +16
    const int kbB = hi * 16;        // B: lanes 16-31 carry K offset +16
    const int n0  = tN * (16 * NT) + lh;

    v8f acc[NT];
#pragma unroll
    for (int nt = 0; nt < NT; ++nt) acc[nt] = v8f{};

    const half_t* aprow = A + (size_t)rowA * lda + kbA;
    const half_t* bprow = B + (size_t)n0 * ldb + kbB;

    for (int kk = 0; kk < K; kk += 32) {
        v16h a;
        v8h a0 = *(const v8h*)(aprow + kk);
        v8h a1 = *(const v8h*)(aprow + kk + 8);
#pragma unroll
        for (int i = 0; i < 8; ++i) { a[i] = a0[i]; a[i + 8] = a1[i]; }
#pragma unroll
        for (int nt = 0; nt < NT; ++nt) {
            const half_t* bp = bprow + (size_t)nt * 16 * ldb + kk;
            v8h b0 = *(const v8h*)(bp);
            v8h b1 = *(const v8h*)(bp + 8);
            v16h b;
#pragma unroll
            for (int i = 0; i < 8; ++i) { b[i] = b0[i]; b[i + 8] = b1[i]; }
            acc[nt] = __builtin_amdgcn_wmma_f32_16x16x32_f16(false, a, false, b,
                                                             (short)0, acc[nt],
                                                             false, false);
        }
    }

#pragma unroll
    for (int nt = 0; nt < NT; ++nt) {
        const int n  = n0 + nt * 16;
        const float bn = bias ? bias[n] : 0.f;
#pragma unroll
        for (int r = 0; r < 8; ++r) {
            const int m = tM * 16 + r + hi * 8;
            float v = acc[nt][r] * alpha + bn;
            const long long idx = coff + (long long)m * ldc + n;
            if (resid)    v += resid[idx];
            if (ACT == 1) v = v / (1.f + __expf(-v));       // silu
            if (rowmask)  v *= rowmask[m] ? 1.f : 0.f;
            if (Cf) Cf[idx] = v;
            if (Ch) Ch[idx] = (half_t)v;
        }
    }
}

// --------------------- weight convert: f32 (KxN) -> f16 transposed (NxK) ----
__global__ void convT_kernel(const float* in, half_t* out, int Kd, int Nd)
{
    int idx = blockIdx.x * 256 + threadIdx.x;
    if (idx < Kd * Nd) {
        int k = idx / Nd, n = idx % Nd;
        out[(size_t)n * Kd + k] = (half_t)in[idx];
    }
}

// ------- f16 transpose: v16 (ROWS x DIM) -> vT16 [b][d][j]  (B,DIM,SEQ) -----
__global__ void vtrans_kernel(const half_t* in, half_t* out)
{
    int idx = blockIdx.x * 256 + threadIdx.x;   // ROWS*DIM total
    int row = idx >> 7;                         // b*SEQ + j
    int d   = idx & 127;
    int b   = row >> 10;
    int j   = row & 1023;
    out[((size_t)b * DIM + d) * SEQ + j] = in[idx];
}

// --------------------------- input embedding + LN ---------------------------
__global__ __launch_bounds__(128)
void embed_kernel(const int* ridx, const int* cidx, const int* pidx,
                  const float* resid, const float* coords, const unsigned char* mask,
                  const float* remb, const float* cemb, const float* pemb,
                  const float* rW, const float* rb,
                  const float* cW, const float* cb,
                  const float* g, const float* b,
                  float* hidden, float* refined)
{
    const int row = blockIdx.x, t = threadIdx.x;
    __shared__ float sb[128];
    __shared__ float stat[2];
    int ri = min(max(ridx[row], 0), 4);
    int ci = min(max(cidx[row], 0), 63);
    int pi = min(max(pidx[row], 0), 63);
    float x = remb[ri * DIM + t] + cemb[ci * DIM + t] + pemb[pi * DIM + t]
            + resid[row] * rW[t] + rb[t]
            + coords[row * 3 + 0] * cW[0 * DIM + t]
            + coords[row * 3 + 1] * cW[1 * DIM + t]
            + coords[row * 3 + 2] * cW[2 * DIM + t] + cb[t];
    sb[t] = x; __syncthreads();
    for (int s = 64; s; s >>= 1) { if (t < s) sb[t] += sb[t + s]; __syncthreads(); }
    if (t == 0) stat[0] = sb[0] / (float)DIM;
    __syncthreads();
    float d = x - stat[0];
    sb[t] = d * d; __syncthreads();
    for (int s = 64; s; s >>= 1) { if (t < s) sb[t] += sb[t + s]; __syncthreads(); }
    if (t == 0) stat[1] = rsqrtf(sb[0] / (float)DIM + LN_EPS);
    __syncthreads();
    float mf = mask[row] ? 1.f : 0.f;
    hidden[row * DIM + t] = (d * stat[1] * g[t] + b[t]) * mf;
    if (t < 3) refined[row * 3 + t] = coords[row * 3 + t];
}

// ------------------------------ LN -> f16 -----------------------------------
__global__ __launch_bounds__(128)
void ln_kernel(const float* in, const float* g, const float* b, half_t* out)
{
    const int row = blockIdx.x, t = threadIdx.x;
    __shared__ float sb[128];
    __shared__ float stat[2];
    float x = in[row * DIM + t];
    sb[t] = x; __syncthreads();
    for (int s = 64; s; s >>= 1) { if (t < s) sb[t] += sb[t + s]; __syncthreads(); }
    if (t == 0) stat[0] = sb[0] / (float)DIM;
    __syncthreads();
    float d = x - stat[0];
    sb[t] = d * d; __syncthreads();
    for (int s = 64; s; s >>= 1) { if (t < s) sb[t] += sb[t + s]; __syncthreads(); }
    if (t == 0) stat[1] = rsqrtf(sb[0] / (float)DIM + LN_EPS);
    __syncthreads();
    out[row * DIM + t] = (half_t)(d * stat[1] * g[t] + b[t]);
}

// -------------- distance-bias MLP, fused add into logits --------------------
__global__ __launch_bounds__(256)
void dbias_kernel(const float* coords, float* logits,
                  const float* W1, const float* b1,
                  const float* W2, const float* b2)
{
    const int i = blockIdx.x, b = blockIdx.y, t = threadIdx.x;
    __shared__ float sw1[DIM], sb1[DIM], sw2[DIM * NH];
    if (t < DIM) { sw1[t] = W1[t]; sb1[t] = b1[t]; }
    for (int u = t; u < DIM * NH; u += 256) sw2[u] = W2[u];
    __syncthreads();
    const float cx = coords[(b * SEQ + i) * 3 + 0];
    const float cy = coords[(b * SEQ + i) * 3 + 1];
    const float cz = coords[(b * SEQ + i) * 3 + 2];
    for (int jj = 0; jj < SEQ / 256; ++jj) {
        const int j = jj * 256 + t;
        float dx = coords[(b * SEQ + j) * 3 + 0] - cx;
        float dy = coords[(b * SEQ + j) * 3 + 1] - cy;
        float dz = coords[(b * SEQ + j) * 3 + 2] - cz;
        float dist = fmaxf(sqrtf(dx * dx + dy * dy + dz * dz), 1e-6f);
        float a0 = b2[0], a1 = b2[1], a2 = b2[2], a3 = b2[3];
#pragma unroll 8
        for (int d = 0; d < DIM; ++d) {
            float tt = dist * sw1[d] + sb1[d];
            float s  = tt / (1.f + __expf(-tt));
            a0 += s * sw2[d * NH + 0];
            a1 += s * sw2[d * NH + 1];
            a2 += s * sw2[d * NH + 2];
            a3 += s * sw2[d * NH + 3];
        }
        const size_t base = (((size_t)b * NH) * SEQ + i) * SEQ + j;
        const size_t hs = (size_t)SEQ * SEQ;
        logits[base + 0 * hs] += a0;
        logits[base + 1 * hs] += a1;
        logits[base + 2 * hs] += a2;
        logits[base + 3 * hs] += a3;
    }
}

// -------- softmax (mask, renorm) in place (f32) + f16 copy for attn@v -------
__global__ __launch_bounds__(256)
void softmax_kernel(float* logits, half_t* attn16, const unsigned char* mask)
{
    const int i = blockIdx.x, bh = blockIdx.y, t = threadIdx.x;
    const int b = bh >> 2;
    float* rowp = logits + ((size_t)bh * SEQ + i) * SEQ;
    half_t* r16 = attn16 + ((size_t)bh * SEQ + i) * SEQ;
    __shared__ float s[256];
    float x[SEQ / 256];
    float mx = -1e30f;
    for (int jj = 0; jj < SEQ / 256; ++jj) {
        int j = jj * 256 + t;
        float v = mask[b * SEQ + j] ? rowp[j] : -10000.f;
        x[jj] = v; mx = fmaxf(mx, v);
    }
    s[t] = mx; __syncthreads();
    for (int st = 128; st; st >>= 1) { if (t < st) s[t] = fmaxf(s[t], s[t + st]); __syncthreads(); }
    const float rowmax = s[0];
    __syncthreads();
    float sum = 0.f;
    for (int jj = 0; jj < SEQ / 256; ++jj) { x[jj] = __expf(x[jj] - rowmax); sum += x[jj]; }
    s[t] = sum; __syncthreads();
    for (int st = 128; st; st >>= 1) { if (t < st) s[t] += s[t + st]; __syncthreads(); }
    const float total = s[0];
    const float mfi = mask[b * SEQ + i] ? 1.f : 0.f;
    const float inv = mfi / fmaxf(total, 1e-30f);   // softmax * mask_i, then renorm
    for (int jj = 0; jj < SEQ / 256; ++jj) {
        int j = jj * 256 + t;
        float a = x[jj] * inv;
        rowp[j] = a;
        r16[j]  = (half_t)a;
    }
}

// -------- coord_delta[b,i,:] = sum_j mean_h(attn) * (coords[j]-coords[i]) ---
__global__ __launch_bounds__(256)
void cdelta_kernel(const float* attn, const float* coords, float* delta)
{
    const int i = blockIdx.x, b = blockIdx.y, t = threadIdx.x;
    const size_t hs = (size_t)SEQ * SEQ;
    const float* a0 = attn + (((size_t)b * NH + 0) * SEQ + i) * SEQ;
    const float* a1 = a0 + hs;
    const float* a2 = a1 + hs;
    const float* a3 = a2 + hs;
    __shared__ float sx[256], sy[256], sz[256], ss[256];
    float vx = 0, vy = 0, vz = 0, vs = 0;
    for (int jj = 0; jj < SEQ / 256; ++jj) {
        int j = jj * 256 + t;
        float am = 0.25f * (a0[j] + a1[j] + a2[j] + a3[j]);
        vx += am * coords[(b * SEQ + j) * 3 + 0];
        vy += am * coords[(b * SEQ + j) * 3 + 1];
        vz += am * coords[(b * SEQ + j) * 3 + 2];
        vs += am;
    }
    sx[t] = vx; sy[t] = vy; sz[t] = vz; ss[t] = vs; __syncthreads();
    for (int st = 128; st; st >>= 1) {
        if (t < st) { sx[t] += sx[t+st]; sy[t] += sy[t+st]; sz[t] += sz[t+st]; ss[t] += ss[t+st]; }
        __syncthreads();
    }
    if (t == 0) {
        const int r = b * SEQ + i;
        delta[r * 3 + 0] = sx[0] - ss[0] * coords[r * 3 + 0];
        delta[r * 3 + 1] = sy[0] - ss[0] * coords[r * 3 + 1];
        delta[r * 3 + 2] = sz[0] - ss[0] * coords[r * 3 + 2];
    }
}

// ----------------- coordinate gate (dot + sigmoid) + update -----------------
__global__ __launch_bounds__(128)
void gate_kernel(const half_t* cga, const float* W2, const float* b2,
                 const float* delta, const unsigned char* mask, float* coords)
{
    const int m = blockIdx.x, t = threadIdx.x;
    __shared__ float s[128];
    __shared__ float gg;
    s[t] = (float)cga[(size_t)m * DIM + t] * W2[t];
    __syncthreads();
    for (int st = 64; st; st >>= 1) { if (t < st) s[t] += s[t + st]; __syncthreads(); }
    if (t == 0) gg = 1.f / (1.f + __expf(-(s[0] + b2[0])));
    __syncthreads();
    if (t < 3) {
        float mf = mask[m] ? 1.f : 0.f;
        coords[m * 3 + t] = (coords[m * 3 + t] + COORD_STEP * gg * delta[m * 3 + t]) * mf;
    }
}

// ===========================================================================
extern "C" void kernel_launch(void* const* d_in, const int* in_sizes, int n_in,
                              void* d_out, int out_size, void* d_ws, size_t ws_size,
                              hipStream_t stream)
{
    (void)in_sizes; (void)n_in; (void)out_size; (void)ws_size;
    const int*   residue_idx = (const int*)d_in[0];
    const int*   chain_idx   = (const int*)d_in[1];
    const int*   copy_idx    = (const int*)d_in[2];
    const float* resid_in    = (const float*)d_in[3];
    const float* coords_in   = (const float*)d_in[4];
    const unsigned char* mask = (const unsigned char*)d_in[5];
    auto F = [&](int i) { return (const float*)d_in[i]; };
    const int LP0 = 15;
    auto LP = [&](int ly, int o) { return (const float*)d_in[LP0 + ly * 24 + o]; };
    // layer offsets: 0 h_g 1 h_b 2 ff_g 3 ff_b 4 Wq 5 bq 6 Wk 7 bk 8 Wv 9 bv
    //  10 Wo 11 bo 12 db_W1 13 db_b1 14 db_W2 15 db_b2 16 cg_W1 17 cg_b1
    //  18 cg_W2 19 cg_b2 20 ff_W1 21 ff_b1 22 ff_W2 23 ff_b2

    // ---- workspace carve-up (256B aligned)
    char* wsb = (char*)d_ws;
    size_t off = 0;
    auto alloc = [&](size_t bytes) { void* p = wsb + off; off = (off + bytes + 255) & ~(size_t)255; return p; };
    float*  hidden  = (float*) alloc((size_t)ROWS * DIM * 4);
    float*  refined = (float*) alloc((size_t)ROWS * 3 * 4);
    float*  delta   = (float*) alloc((size_t)ROWS * 3 * 4);
    half_t* h16     = (half_t*)alloc((size_t)ROWS * DIM * 2);
    half_t* q16     = (half_t*)alloc((size_t)ROWS * DIM * 2);
    half_t* k16     = (half_t*)alloc((size_t)ROWS * DIM * 2);
    half_t* v16     = (half_t*)alloc((size_t)ROWS * DIM * 2);
    half_t* vT16    = (half_t*)alloc((size_t)ROWS * DIM * 2);   // [b][d][j]
    half_t* msg16   = (half_t*)alloc((size_t)ROWS * DIM * 2);
    half_t* hid16   = (half_t*)alloc((size_t)ROWS * DIM * 2);
    half_t* cga16   = (half_t*)alloc((size_t)ROWS * DIM * 2);
    half_t* ffa16   = (half_t*)alloc((size_t)ROWS * FFD * 2);
    float*  logits  = (float*) alloc((size_t)BATCH * NH * SEQ * SEQ * 4);  // 32MB, L2-resident
    half_t* attn16  = (half_t*)alloc((size_t)BATCH * NH * SEQ * SEQ * 2);  // 16MB
    half_t* WqT[2], *WkT[2], *WvT[2], *WoT[2], *cgW1T[2], *ffW1T[2], *ffW2T[2];
    for (int ly = 0; ly < 2; ++ly) {
        WqT[ly]   = (half_t*)alloc((size_t)DIM * DIM * 2);
        WkT[ly]   = (half_t*)alloc((size_t)DIM * DIM * 2);
        WvT[ly]   = (half_t*)alloc((size_t)DIM * DIM * 2);
        WoT[ly]   = (half_t*)alloc((size_t)DIM * DIM * 2);
        cgW1T[ly] = (half_t*)alloc((size_t)DIM * DIM * 2);
        ffW1T[ly] = (half_t*)alloc((size_t)DIM * FFD * 2);
        ffW2T[ly] = (half_t*)alloc((size_t)FFD * DIM * 2);
    }

    auto convT = [&](const float* in, half_t* out, int K, int N) {
        int n = K * N;
        convT_kernel<<<(n + 255) / 256, 256, 0, stream>>>(in, out, K, N);
    };
    // NT selected by N (all N here are 32 or multiples of 64)
    auto gemm = [&](const half_t* A, int lda, long long sAo, long long sAi,
                    const half_t* B, int ldb, long long sBo, long long sBi,
                    const float* bias, float alpha, const float* residp,
                    float* Cf, half_t* Ch, int ldc, long long sCo, long long sCi,
                    const unsigned char* rmask,
                    int M, int N, int K, int zc, int zdiv, int act) {
        if (N % 64 == 0) {
            dim3 g(M / 16, N / 64, zc);
            if (act == 0)
                gemm_wmma_kernel<4, 0><<<g, 32, 0, stream>>>(A, lda, sAo, sAi, B, ldb, sBo, sBi,
                    bias, alpha, residp, Cf, Ch, ldc, sCo, sCi, rmask, zdiv, K);
            else
                gemm_wmma_kernel<4, 1><<<g, 32, 0, stream>>>(A, lda, sAo, sAi, B, ldb, sBo, sBi,
                    bias, alpha, residp, Cf, Ch, ldc, sCo, sCi, rmask, zdiv, K);
        } else {  // N == 32
            dim3 g(M / 16, N / 32, zc);
            gemm_wmma_kernel<2, 0><<<g, 32, 0, stream>>>(A, lda, sAo, sAi, B, ldb, sBo, sBi,
                bias, alpha, residp, Cf, Ch, ldc, sCo, sCi, rmask, zdiv, K);
        }
    };

    // ---- convert all WMMA weights to transposed f16 (recomputed every call)
    for (int ly = 0; ly < 2; ++ly) {
        convT(LP(ly, 4),  WqT[ly],   DIM, DIM);
        convT(LP(ly, 6),  WkT[ly],   DIM, DIM);
        convT(LP(ly, 8),  WvT[ly],   DIM, DIM);
        convT(LP(ly, 10), WoT[ly],   DIM, DIM);
        convT(LP(ly, 16), cgW1T[ly], DIM, DIM);
        convT(LP(ly, 20), ffW1T[ly], DIM, FFD);
        convT(LP(ly, 22), ffW2T[ly], FFD, DIM);
    }

    // ---- embedding + input LN, init refined coords
    embed_kernel<<<ROWS, 128, 0, stream>>>(residue_idx, chain_idx, copy_idx,
        resid_in, coords_in, mask, F(6), F(7), F(8), F(9), F(10), F(11), F(12),
        F(13), F(14), hidden, refined);

    const long long HL = (long long)SEQ * SEQ;       // per-(b,h) attn stride
    const long long BL = (long long)SEQ * DIM;       // per-b row-block stride
    const float inv_sqrt_dh = 0.17677669529663687f;  // 1/sqrt(32)

    for (int rc = 0; rc < 2; ++rc) {
        for (int ly = 0; ly < 2; ++ly) {
            // h = LN(hidden)
            ln_kernel<<<ROWS, 128, 0, stream>>>(hidden, LP(ly, 0), LP(ly, 1), h16);
            // q,k,v = h @ W + b  (f16 out)
            gemm(h16, DIM, 0, 0, WqT[ly], DIM, 0, 0, LP(ly, 5), 1.f, nullptr,
                 nullptr, q16, DIM, 0, 0, nullptr, ROWS, DIM, DIM, 1, 1, 0);
            gemm(h16, DIM, 0, 0, WkT[ly], DIM, 0, 0, LP(ly, 7), 1.f, nullptr,
                 nullptr, k16, DIM, 0, 0, nullptr, ROWS, DIM, DIM, 1, 1, 0);
            gemm(h16, DIM, 0, 0, WvT[ly], DIM, 0, 0, LP(ly, 9), 1.f, nullptr,
                 nullptr, v16, DIM, 0, 0, nullptr, ROWS, DIM, DIM, 1, 1, 0);
            // v transposed to [b][d][j] so attn@v uses the contiguous Bt path
            vtrans_kernel<<<(ROWS * DIM) / 256, 256, 0, stream>>>(v16, vT16);
            // logits[b,h] = q_bh @ k_bh^T / sqrt(dh)   (z = b*4+h, zdiv=4)
            // Bt = k16 itself: Bt[n=j][k=d] = k[j][d]
            gemm(q16, DIM, BL, DH, k16, DIM, BL, DH, nullptr, inv_sqrt_dh,
                 nullptr, logits, nullptr, SEQ, (long long)NH * HL, HL, nullptr,
                 SEQ, SEQ, DH, BATCH * NH, NH, 0);
            // + distance-bias MLP (fused, heavy VALU part)
            dbias_kernel<<<dim3(SEQ, BATCH), 256, 0, stream>>>(refined, logits,
                LP(ly, 12), LP(ly, 13), LP(ly, 14), LP(ly, 15));
            // softmax + mask + renorm -> attn (f32 in place) + attn16
            softmax_kernel<<<dim3(SEQ, BATCH * NH), 256, 0, stream>>>(logits, attn16, mask);
            // coord_delta from attn mean over heads and OLD coords
            cdelta_kernel<<<dim3(SEQ, BATCH), 256, 0, stream>>>(logits, refined, delta);
            // msg[b,:,h*32:(h+1)*32] = attn_bh @ v_bh : Bt = vT16[b][h*32+d][j]
            gemm(attn16, SEQ, (long long)NH * HL, HL,
                 vT16, SEQ, (long long)DIM * SEQ, (long long)DH * SEQ,
                 nullptr, 1.f, nullptr, nullptr, msg16, DIM, BL, DH, nullptr,
                 SEQ, DH, SEQ, BATCH * NH, NH, 0);
            // hidden += msg @ Wo + bo
            gemm(msg16, DIM, 0, 0, WoT[ly], DIM, 0, 0, LP(ly, 11), 1.f, hidden,
                 hidden, nullptr, DIM, 0, 0, nullptr, ROWS, DIM, DIM, 1, 1, 0);
            // FF: fh = LN(hidden); hidden = (hidden + silu(fh@W1+b1)@W2 + b2)*mask
            ln_kernel<<<ROWS, 128, 0, stream>>>(hidden, LP(ly, 2), LP(ly, 3), h16);
            gemm(h16, DIM, 0, 0, ffW1T[ly], DIM, 0, 0, LP(ly, 21), 1.f, nullptr,
                 nullptr, ffa16, FFD, 0, 0, nullptr, ROWS, FFD, DIM, 1, 1, 1);
            gemm(ffa16, FFD, 0, 0, ffW2T[ly], FFD, 0, 0, LP(ly, 23), 1.f, hidden,
                 hidden, hid16, DIM, 0, 0, mask, ROWS, DIM, FFD, 1, 1, 0);
            // coord gate: silu(hidden@cgW1+b1) (f16), then dot cgW2 + sigmoid
            gemm(hid16, DIM, 0, 0, cgW1T[ly], DIM, 0, 0, LP(ly, 17), 1.f, nullptr,
                 nullptr, cga16, DIM, 0, 0, nullptr, ROWS, DIM, DIM, 1, 1, 1);
            gate_kernel<<<ROWS, 128, 0, stream>>>(cga16, LP(ly, 18), LP(ly, 19),
                                                  delta, mask, refined);
        }
    }

    hipMemcpyAsync(d_out, refined, (size_t)ROWS * 3 * sizeof(float),
                   hipMemcpyDeviceToDevice, stream);
}
